// RootPersistencePressure_88304527606107
// MI455X (gfx1250) — 16-bit and 8-bit matrix operand order repaired and checked by reading, compiled, not run on hardware
//
#include <hip/hip_runtime.h>

#define N_SAMP 100000
#define N_PERS 99999
#define NPAD   100096
#define KDIM   512
#define MDIM   256
#define PT_BYTES ((size_t)MDIM * (size_t)NPAD)
#define TH_ELEMS (MDIM * KDIM)
#define OUT_ELEMS (MDIM * (MDIM - 1) / 2)

typedef __attribute__((ext_vector_type(8)))  float          v8f;
typedef __attribute__((ext_vector_type(16))) __bf16         v16bf;
typedef __attribute__((ext_vector_type(16))) unsigned short v16u;
typedef __attribute__((ext_vector_type(8)))  unsigned int   v8ui;
typedef __attribute__((ext_vector_type(16))) int            v16i;
typedef __attribute__((ext_vector_type(2)))  int            v2i;
typedef __attribute__((ext_vector_type(4)))  int            v4i;

union BFVec { v16u u; v8ui w; v16bf b; };

struct PackPair { unsigned hi; unsigned lo; };

__device__ __forceinline__ unsigned short f2bf(float f) {
  unsigned u = __float_as_uint(f);
  u += 0x7FFFu + ((u >> 16) & 1u);
  return (unsigned short)(u >> 16);
}
__device__ __forceinline__ float bf2f(unsigned short h) {
  return __uint_as_float(((unsigned)h) << 16);
}
__device__ __forceinline__ PackPair split_pack(float f0, float f1) {
  unsigned short h0 = f2bf(f0), h1 = f2bf(f1);
  PackPair p;
  p.hi = (unsigned)h0 | ((unsigned)h1 << 16);
  p.lo = (unsigned)f2bf(f0 - bf2f(h0)) | ((unsigned)f2bf(f1 - bf2f(h1)) << 16);
  return p;
}

#define WMMA_BF16(A, B, C) \
  __builtin_amdgcn_wmma_f32_16x16x32_bf16(false, (A), false, (B), (short)0, (C), false, false)

__global__ void prep_kernel(const float* __restrict__ theta,
                            const float* __restrict__ noise,
                            unsigned short* __restrict__ th_hi,
                            unsigned short* __restrict__ th_lo,
                            unsigned short* __restrict__ tp_hi,
                            unsigned short* __restrict__ tp_lo,
                            unsigned char* __restrict__ PT,
                            float* __restrict__ gram) {
  int idx = blockIdx.x * blockDim.x + threadIdx.x;
  if (idx < TH_ELEMS) {
    float t  = theta[idx];
    float tp = fmaf(noise[idx], 0.01f, t);
    unsigned short h0 = f2bf(t);
    th_hi[idx] = h0;
    th_lo[idx] = f2bf(t - bf2f(h0));
    unsigned short h1 = f2bf(tp);
    tp_hi[idx] = h1;
    tp_lo[idx] = f2bf(tp - bf2f(h1));
  }
  if (idx < MDIM * MDIM) gram[idx] = 0.0f;
  if (idx < MDIM * (NPAD - N_SAMP)) {
    int m  = idx / (NPAD - N_SAMP);
    int nn = idx % (NPAD - N_SAMP);
    PT[(unsigned)m * NPAD + N_SAMP + nn] = 0;
  }
}

__global__ __launch_bounds__(32)
void phi_pers_kernel(const char* __restrict__ basis,
                     const char* __restrict__ th_hi,
                     const char* __restrict__ th_lo,
                     const char* __restrict__ tp_hi,
                     const char* __restrict__ tp_lo,
                     unsigned char* __restrict__ PT) {
  const int lane = threadIdx.x;
  const int l16  = lane & 15;
  const bool hh  = lane >= 16;
  const int ca   = hh ? 16 : 0;
  const int cb   = hh ? 16 : 0;
  const int t    = blockIdx.x;
  const int g    = blockIdx.y;

  int row0 = 16 * t + l16;
  int row1 = row0 + 1;
  if (row1 > N_SAMP - 1) row1 = N_SAMP - 1;

  unsigned aoff0 = ((unsigned)row0 * KDIM + ca) * 4u;
  unsigned aoff1 = ((unsigned)row1 * KDIM + ca) * 4u;
  unsigned boffm = ((unsigned)(g * 64 + l16) * KDIM + cb) * 2u;

  v8f acc[4][4];
#pragma unroll
  for (int a = 0; a < 4; ++a)
#pragma unroll
    for (int b = 0; b < 4; ++b)
#pragma unroll
      for (int e = 0; e < 8; ++e) acc[a][b][e] = 0.0f;

  for (int kb = 0; kb < KDIM; kb += 32) {
    v8f fa0 = *(const v8f*)(basis + aoff0);
    v8f fb0 = *(const v8f*)(basis + aoff0 + 32);
    v8f fa1 = *(const v8f*)(basis + aoff1);
    v8f fb1 = *(const v8f*)(basis + aoff1 + 32);
    aoff0 += 32 * 4;
    aoff1 += 32 * 4;
    BFVec a0h, a0l, a1h, a1l;
#pragma unroll
    for (int e = 0; e < 4; ++e) {
      PackPair p;
      p = split_pack(fa0[2 * e], fa0[2 * e + 1]);
      a0h.w[e] = p.hi;      a0l.w[e] = p.lo;
      p = split_pack(fb0[2 * e], fb0[2 * e + 1]);
      a0h.w[4 + e] = p.hi;  a0l.w[4 + e] = p.lo;
      p = split_pack(fa1[2 * e], fa1[2 * e + 1]);
      a1h.w[e] = p.hi;      a1l.w[e] = p.lo;
      p = split_pack(fb1[2 * e], fb1[2 * e + 1]);
      a1h.w[4 + e] = p.hi;  a1l.w[4 + e] = p.lo;
    }
#pragma unroll
    for (int mt = 0; mt < 4; ++mt) {
      const unsigned bo = boffm + (unsigned)(mt * 16 * KDIM * 2) +
                          (unsigned)kb * 2u;
      BFVec bhr, blr, bhp, blp;
      bhr.u = *(const v16u*)(th_hi + bo);
      blr.u = *(const v16u*)(th_lo + bo);
      bhp.u = *(const v16u*)(tp_hi + bo);
      blp.u = *(const v16u*)(tp_lo + bo);
      acc[mt][0] = WMMA_BF16(a0h.b, bhr.b, acc[mt][0]);
      acc[mt][0] = WMMA_BF16(a0h.b, blr.b, acc[mt][0]);
      acc[mt][0] = WMMA_BF16(a0l.b, bhr.b, acc[mt][0]);
      acc[mt][1] = WMMA_BF16(a1h.b, bhr.b, acc[mt][1]);
      acc[mt][1] = WMMA_BF16(a1h.b, blr.b, acc[mt][1]);
      acc[mt][1] = WMMA_BF16(a1l.b, bhr.b, acc[mt][1]);
      acc[mt][2] = WMMA_BF16(a0h.b, bhp.b, acc[mt][2]);
      acc[mt][2] = WMMA_BF16(a0h.b, blp.b, acc[mt][2]);
      acc[mt][2] = WMMA_BF16(a0l.b, bhp.b, acc[mt][2]);
      acc[mt][3] = WMMA_BF16(a1h.b, bhp.b, acc[mt][3]);
      acc[mt][3] = WMMA_BF16(a1h.b, blp.b, acc[mt][3]);
      acc[mt][3] = WMMA_BF16(a1l.b, bhp.b, acc[mt][3]);
    }
  }

  const int nbase = 16 * t + (hh ? 8 : 0);
#pragma unroll
  for (int mt = 0; mt < 4; ++mt) {
    unsigned long long q = 0ull;
#pragma unroll
    for (int r = 0; r < 8; ++r) {
      bool pers = (acc[mt][0][r] * acc[mt][1][r] < 0.0f) &&
                  (acc[mt][2][r] * acc[mt][3][r] < 0.0f) &&
                  (nbase + r < N_PERS);
      if (pers) q |= (0x38ull << (8 * r));
    }
    const unsigned mrow = (unsigned)(g * 64 + mt * 16 + l16);
    *(unsigned long long*)(PT + mrow * (unsigned)NPAD + (unsigned)nbase) = q;
  }
}

#define KCHUNK (98 * 128)

__global__ __launch_bounds__(32)
void gram_kernel(const char* __restrict__ PT,
                 float* __restrict__ gram) {
  const int lane = threadIdx.x;
  const int l16  = lane & 15;
  const bool hh  = lane >= 16;
  const unsigned ca = hh ? 16u : 0u;
  const unsigned cb = hh ? 16u : 0u;

  int p = blockIdx.x, ti = 0;
  while (p >= 16 - ti) { p -= 16 - ti; ++ti; }
  const int tj = ti + p;

  const unsigned kb0 = blockIdx.y * (unsigned)KCHUNK;
  unsigned kb1 = kb0 + KCHUNK;
  if (kb1 > NPAD) kb1 = NPAD;

  unsigned aoff = (unsigned)(ti * 16 + l16) * (unsigned)NPAD + kb0 + ca;
  unsigned boff = (unsigned)(tj * 16 + l16) * (unsigned)NPAD + kb0 + cb;

  v8f acc;
#pragma unroll
  for (int e = 0; e < 8; ++e) acc[e] = 0.0f;

  for (unsigned kb = kb0; kb < kb1; kb += 128) {
    v16i A, B;
#pragma unroll
    for (int ch = 0; ch < 8; ++ch) {
      v2i q = *(const v2i*)(PT + aoff + (ch >> 1) * 32 + (ch & 1) * 8);
      A[2 * ch]     = q[0];
      A[2 * ch + 1] = q[1];
    }
#pragma unroll
    for (int ch = 0; ch < 4; ++ch) {
      v4i q = *(const v4i*)(PT + boff + ch * 32);
      B[4 * ch]     = q[0];
      B[4 * ch + 1] = q[1];
      B[4 * ch + 2] = q[2];
      B[4 * ch + 3] = q[3];
    }
    aoff += 128;
    boff += 128;
    acc = __builtin_amdgcn_wmma_f32_16x16x128_fp8_fp8(A, B, (short)0, acc,
                                                      false, false);
  }

  const int j     = tj * 16 + l16;
  const int ibase = ti * 16 + (hh ? 8 : 0);
#pragma unroll
  for (int r = 0; r < 8; ++r) {
    atomicAdd(gram + (ibase + r) * MDIM + j, acc[r]);
  }
}

__global__ void gather_kernel(const float* __restrict__ gram,
                              float* __restrict__ out) {
  int idx = blockIdx.x * blockDim.x + threadIdx.x;
  if (idx >= OUT_ELEMS) return;
  int i = (int)((511.0 - sqrt(511.0 * 511.0 - 8.0 * (double)idx)) * 0.5);
  if (i < 0) i = 0;
  if (i > MDIM - 2) i = MDIM - 2;
  while (i > 0 && (i * 255 - i * (i - 1) / 2) > idx) --i;
  while (((i + 1) * 255 - (i + 1) * i / 2) <= idx) ++i;
  int j = i + 1 + (idx - (i * 255 - i * (i - 1) / 2));
  out[idx] = gram[i * MDIM + j];
}

extern "C" void kernel_launch(void* const* d_in, const int* in_sizes, int n_in,
                              void* d_out, int out_size, void* d_ws,
                              size_t ws_size, hipStream_t stream) {
  (void)in_sizes; (void)n_in; (void)out_size; (void)ws_size;
  const float* theta = (const float*)d_in[0];
  const float* basis = (const float*)d_in[1];
  const float* noise = (const float*)d_in[2];
  float* out = (float*)d_out;

  char* ws = (char*)d_ws;
  unsigned char*  PT    = (unsigned char*)ws;
  unsigned short* th_hi = (unsigned short*)(ws + PT_BYTES);
  unsigned short* th_lo = th_hi + TH_ELEMS;
  unsigned short* tp_hi = th_lo + TH_ELEMS;
  unsigned short* tp_lo = tp_hi + TH_ELEMS;
  float*          gram  = (float*)(tp_lo + TH_ELEMS);

  prep_kernel<<<(TH_ELEMS + 255) / 256, 256, 0, stream>>>(
      theta, noise, th_hi, th_lo, tp_hi, tp_lo, PT, gram);

  phi_pers_kernel<<<dim3(N_SAMP / 16, 4), 32, 0, stream>>>(
      (const char*)basis, (const char*)th_hi, (const char*)th_lo,
      (const char*)tp_hi, (const char*)tp_lo, PT);

  gram_kernel<<<dim3(136, 8), 32, 0, stream>>>((const char*)PT, gram);

  gather_kernel<<<(OUT_ELEMS + 255) / 256, 256, 0, stream>>>(gram, out);
}
